// GCNRec_90812788506714
// MI455X (gfx1250) — matrix operand order repaired and checked by reading, not compiled, not run on hardware
//
#include <hip/hip_runtime.h>
#include <hip/hip_bf16.h>
#include <math.h>

typedef __attribute__((ext_vector_type(16))) __bf16 v16bf;
typedef __attribute__((ext_vector_type(8)))  float  v8f;
typedef __attribute__((ext_vector_type(4)))  int    i32x4;

#define NB_OTHER 2000
#define NB_USER  20000
#define NB_ITEM  50000
#define NNODES   (NB_OTHER + NB_USER + NB_ITEM)   /* 72000 */
#define EMB      64
#define KER      128
#define TT       20
#define VOCAB1   50001
#define NEDGE    (NNODES * 16)
#define BB       1024
#define KK       4
#define NGATHER  (BB + BB + KK * BB)              /* 6144 */

#if defined(__HIP_DEVICE_COMPILE__) && __has_builtin(__builtin_amdgcn_global_load_async_to_lds_b128)
#define USE_ASYNC_LDS 1
#else
#define USE_ASYNC_LDS 0
#endif

// Address-space-qualified pointee types for the async builtin
// (probe-confirmed prototype: (int4 AS1*, int4 AS3*, imm int, imm int)).
typedef __attribute__((address_space(1))) i32x4 as1_i32x4;
typedef __attribute__((address_space(3))) i32x4 as3_i32x4;

union FragB16 { v16bf v; uint4 u[2]; __bf16 e[16]; };

static __device__ inline v8f wmma_bf16(v16bf a, v16bf b, v8f c) {
  // 8 args: (neg_a, A, neg_b, B, c_mod, C, reuse_a, reuse_b)
  return __builtin_amdgcn_wmma_f32_16x16x32_bf16(false, a, false, b, (short)0, c,
                                                 false, false);
}

// Fast transcendentals: CDNA5 has hardware v_tanh_f32; fall back to v_exp+v_rcp.
static __device__ inline float fast_tanh(float x) {
#if defined(__HIP_DEVICE_COMPILE__) && __has_builtin(__builtin_amdgcn_tanhf)
  return __builtin_amdgcn_tanhf(x);
#elif defined(__HIP_DEVICE_COMPILE__) && __has_builtin(__builtin_amdgcn_tanh_f32)
  return __builtin_amdgcn_tanh_f32(x);
#else
  return 1.0f - 2.0f * __builtin_amdgcn_rcpf(__expf(2.0f * x) + 1.0f);
#endif
}
static __device__ inline float fast_sigmoid(float x) {
#if defined(__HIP_DEVICE_COMPILE__) && (__has_builtin(__builtin_amdgcn_tanhf) || __has_builtin(__builtin_amdgcn_tanh_f32))
  return 0.5f * fast_tanh(0.5f * x) + 0.5f;
#else
  return __builtin_amdgcn_rcpf(1.0f + __expf(-x));
#endif
}

#if USE_ASYNC_LDS
static __device__ inline void wait_async_zero() {
#if __has_builtin(__builtin_amdgcn_s_wait_asynccnt)
  __builtin_amdgcn_s_wait_asynccnt(0);
#else
  asm volatile("s_wait_asynccnt 0x0" ::: "memory");
#endif
}
static __device__ inline void async_copy_b128(const __bf16* gsrc, __bf16* ldst) {
  __builtin_amdgcn_global_load_async_to_lds_b128((as1_i32x4*)gsrc, (as3_i32x4*)ldst, 0, 0);
}
#endif

// A fragment (16x32 bf16, K-chunk kc) from a row-major bf16 matrix, row stride ld.
// Lane layout: lanes 0-15 row M=lane K[0..15];
// lanes 16-31 same row set K[16..31] (contiguous, matches B loaders).
static __device__ inline v16bf load_a_frag(const __bf16* base, int ld, int kc, int lane) {
  const int half = lane >> 4, row = lane & 15;
  const __bf16* p = base + row * ld + kc * 32 + half * 16;
  FragB16 f;
  f.u[0] = *(const uint4*)p;
  f.u[1] = *(const uint4*)(p + 8);
  return f.v;
}

// B fragment where logical B[k][n] = W[n][k] (W row-major [N][K], f32 -> bf16 cvt).
static __device__ inline v16bf load_b_frag_rowmajor_f32(const float* W, int nbase,
                                                        int ld, int kc, int lane) {
  const int half = lane >> 4, n = lane & 15;
  const float* p = W + (size_t)(nbase + n) * ld + kc * 32 + half * 16;
  FragB16 f;
#pragma unroll
  for (int i = 0; i < 16; ++i) f.e[i] = (__bf16)p[i];
  return f.v;
}

// B fragment where logical B[k][n] = W[k][n] (W row-major [K][N], f32 -> bf16 cvt).
static __device__ inline v16bf load_b_frag_colmajor_f32(const float* W, int nbase,
                                                        int ld, int kc, int lane) {
  const int half = lane >> 4, n = lane & 15;
  const float* p = W + (size_t)(kc * 32 + half * 16) * ld + nbase + n;
  FragB16 f;
#pragma unroll
  for (int i = 0; i < 16; ++i) f.e[i] = (__bf16)p[i * ld];
  return f.v;
}

__global__ void cvt_bf16_kernel(const float* __restrict__ src, __bf16* __restrict__ dst, int n) {
  int i = blockIdx.x * blockDim.x + threadIdx.x;
  if (i < n) dst[i] = (__bf16)src[i];
}

// ---------------------------------------------------------------------------
// Fused embedding-gather + 2-layer GRU + (pos + h) epilogue.
// 1 block = 4 wave32 = one 16-node tile. Wave w owns h columns [16w, 16w+16).
// Layer-0 input gather = async b128 copies into double-buffered LDS, issued one
// step ahead so global latency hides under the 12 WMMAs of the current step.
// ---------------------------------------------------------------------------
__global__ __launch_bounds__(128) void gru_fused_kernel(
    const __bf16* __restrict__ wembbf, const int* __restrict__ lookup,
    const float* __restrict__ wih0, const float* __restrict__ whh0,
    const float* __restrict__ bih0, const float* __restrict__ bhh0,
    const float* __restrict__ wih1, const float* __restrict__ whh1,
    const float* __restrict__ bih1, const float* __restrict__ bhh1,
    const float* __restrict__ other_pos, const float* __restrict__ user_pos,
    const float* __restrict__ item_pos, __bf16* __restrict__ allemb)
{
  __shared__ __align__(16) __bf16 Xbuf[2][16 * EMB];       // double-buffered input tile
  __shared__ __align__(16) __bf16 Hs[16 * EMB];            // hidden state (bf16)
  __shared__ __align__(16) __bf16 Ybuf[TT * 16 * EMB];     // layer-0 output sequence

  const int tid  = threadIdx.x;
  const int wave = tid >> 5, lane = tid & 31;
  const int half = lane >> 4, ln = lane & 15;
  const int mbase = blockIdx.x * 16;
  const int jc = wave * 16;

  // gather assignment: each of 128 threads copies one 16B chunk (16 rows x 8 chunks)
  const int grow = tid >> 3, gcol = (tid & 7) * 8;

#if USE_ASYNC_LDS
  {   // prefetch step 0
    int idx = lookup[(mbase + grow) * TT + 0];
    async_copy_b128(wembbf + (size_t)idx * EMB + gcol, &Xbuf[0][grow * EMB + gcol]);
  }
#endif

  v8f hreg;
  for (int layer = 0; layer < 2; ++layer) {
    const float* wih = layer ? wih1 : wih0;
    const float* whh = layer ? whh1 : whh0;
    const float* bih = layer ? bih1 : bih0;
    const float* bhh = layer ? bhh1 : bhh0;

    __syncthreads();                         // previous layer's LDS writes done
    for (int i = tid; i < 16 * EMB; i += 128) Hs[i] = (__bf16)0.0f;
#pragma unroll
    for (int v = 0; v < 8; ++v) hreg[v] = 0.0f;

    // Weight B-fragments stay resident in VGPRs for all 20 steps.
    v16bf Bih[6], Bhh[6];
#pragma unroll
    for (int g = 0; g < 3; ++g)
#pragma unroll
      for (int c = 0; c < 2; ++c) {
        Bih[g * 2 + c] = load_b_frag_rowmajor_f32(wih, g * EMB + jc, EMB, c, lane);
        Bhh[g * 2 + c] = load_b_frag_rowmajor_f32(whh, g * EMB + jc, EMB, c, lane);
      }
    float bi[3], bh[3];
#pragma unroll
    for (int g = 0; g < 3; ++g) {
      bi[g] = bih[g * EMB + jc + ln];
      bh[g] = bhh[g * EMB + jc + ln];
    }

    for (int t = 0; t < TT; ++t) {
#if USE_ASYNC_LDS
      if (layer == 0) wait_async_zero();     // my step-t tile landed in LDS
      __syncthreads();                       // everyone's tile + h_{t-1} visible
      if (layer == 0 && t + 1 < TT) {        // kick off step t+1 gather
        int idx = lookup[(mbase + grow) * TT + t + 1];
        async_copy_b128(wembbf + (size_t)idx * EMB + gcol,
                        &Xbuf[(t + 1) & 1][grow * EMB + gcol]);
      }
#else
      if (layer == 0) {
        int idx = lookup[(mbase + grow) * TT + t];
        *(uint4*)(&Xbuf[t & 1][grow * EMB + gcol]) =
            *(const uint4*)(wembbf + (size_t)idx * EMB + gcol);
      }
      __syncthreads();
#endif
      const __bf16* Xsrc = (layer == 0) ? Xbuf[t & 1] : (Ybuf + t * 16 * EMB);
      v16bf Ax0 = load_a_frag(Xsrc, EMB, 0, lane);
      v16bf Ax1 = load_a_frag(Xsrc, EMB, 1, lane);
      v16bf Ah0 = load_a_frag(Hs, EMB, 0, lane);
      v16bf Ah1 = load_a_frag(Hs, EMB, 1, lane);

      v8f gi[3], gh[3];
#pragma unroll
      for (int g = 0; g < 3; ++g) {
        v8f a, b;
#pragma unroll
        for (int v = 0; v < 8; ++v) { a[v] = bi[g]; b[v] = bh[g]; }
        a = wmma_bf16(Ax0, Bih[g * 2 + 0], a);
        a = wmma_bf16(Ax1, Bih[g * 2 + 1], a);
        b = wmma_bf16(Ah0, Bhh[g * 2 + 0], b);
        b = wmma_bf16(Ah1, Bhh[g * 2 + 1], b);
        gi[g] = a; gh[g] = b;
      }
      // C/D layout: lane column = jc+ln, VGPR v = row v + 8*half -> per-lane math
      v8f hnew;
#pragma unroll
      for (int v = 0; v < 8; ++v) {
        float r = fast_sigmoid(gi[0][v] + gh[0][v]);
        float z = fast_sigmoid(gi[1][v] + gh[1][v]);
        float n = fast_tanh(gi[2][v] + r * gh[2][v]);
        hnew[v] = (1.0f - z) * n + z * hreg[v];
      }
      hreg = hnew;
      __syncthreads();                       // all waves finished reading Hs
#pragma unroll
      for (int v = 0; v < 8; ++v) {
        int r = v + half * 8;
        __bf16 hb = (__bf16)hnew[v];
        Hs[r * EMB + jc + ln] = hb;
        if (layer == 0) Ybuf[(t * 16 + r) * EMB + jc + ln] = hb;
      }
    }
  }
  // epilogue: all_emb = positional + final hidden (stored bf16 for conv1 WMMA)
#pragma unroll
  for (int v = 0; v < 8; ++v) {
    int node = mbase + v + half * 8;
    int col = jc + ln;
    float p;
    if (node < NB_OTHER)                p = other_pos[(size_t)node * EMB + col];
    else if (node < NB_OTHER + NB_USER) p = user_pos[(size_t)(node - NB_OTHER) * EMB + col];
    else                                p = item_pos[(size_t)(node - NB_OTHER - NB_USER) * EMB + col];
    allemb[(size_t)node * EMB + col] = (__bf16)(p + hreg[v]);
  }
}

// C[n][128] = Abf[n][64] @ W[64][128], A already bf16 in global.
__global__ __launch_bounds__(256) void gemm64_kernel(
    const __bf16* __restrict__ A, const float* __restrict__ W, float* __restrict__ C)
{
  const int tid = threadIdx.x;
  const int wave = tid >> 5, lane = tid & 31;
  const int half = lane >> 4, ln = lane & 15;
  const int mbase = blockIdx.x * 16;
  const int nb = wave * 16;
  v16bf a0 = load_a_frag(A + (size_t)mbase * EMB, EMB, 0, lane);
  v16bf a1 = load_a_frag(A + (size_t)mbase * EMB, EMB, 1, lane);
  v16bf b0 = load_b_frag_colmajor_f32(W, nb, KER, 0, lane);
  v16bf b1 = load_b_frag_colmajor_f32(W, nb, KER, 1, lane);
  v8f acc;
#pragma unroll
  for (int v = 0; v < 8; ++v) acc[v] = 0.0f;
  acc = wmma_bf16(a0, b0, acc);
  acc = wmma_bf16(a1, b1, acc);
#pragma unroll
  for (int v = 0; v < 8; ++v)
    C[(size_t)(mbase + v + half * 8) * KER + nb + ln] = acc[v];
}

// C[n][128] = bf16(X[n][128]) @ W[128][128]; A tile staged f32->bf16 through LDS.
__global__ __launch_bounds__(256) void gemm128_kernel(
    const float* __restrict__ X, const float* __restrict__ W, float* __restrict__ C)
{
  __shared__ __align__(16) __bf16 As[16 * KER];
  const int tid = threadIdx.x;
  const int wave = tid >> 5, lane = tid & 31;
  const int half = lane >> 4, ln = lane & 15;
  const int mbase = blockIdx.x * 16;
  const int nb = wave * 16;
  for (int i = tid; i < 16 * KER; i += 256) {
    int r = i >> 7, c = i & 127;
    As[i] = (__bf16)X[(size_t)(mbase + r) * KER + c];
  }
  __syncthreads();
  v8f acc;
#pragma unroll
  for (int v = 0; v < 8; ++v) acc[v] = 0.0f;
#pragma unroll
  for (int c = 0; c < 4; ++c) {
    v16bf a = load_a_frag(As, KER, c, lane);
    v16bf b = load_b_frag_colmajor_f32(W, nb, KER, c, lane);
    acc = wmma_bf16(a, b, acc);
  }
#pragma unroll
  for (int v = 0; v < 8; ++v)
    C[(size_t)(mbase + v + half * 8) * KER + nb + ln] = acc[v];
}

__global__ void init_bias_kernel(float* __restrict__ Y, const float* __restrict__ b, int n) {
  int i = blockIdx.x * blockDim.x + threadIdx.x;
  if (i < n) Y[i] = b[i & (KER - 1)];
}

// y[row[e]] += vals[e] * x[col[e]]  (random rows -> f32 global atomics), 1 wave/edge
__global__ void spmm_atomic_kernel(const int* __restrict__ row, const int* __restrict__ col,
                                   const float* __restrict__ vals, const float* __restrict__ X,
                                   float* __restrict__ Y, int ne)
{
  const int lane = threadIdx.x & 31;
  int w = (blockIdx.x * blockDim.x + threadIdx.x) >> 5;
  const int nw = (gridDim.x * blockDim.x) >> 5;
  for (int e = w; e < ne; e += nw) {
    int r = row[e], c = col[e];
    float v = vals[e];
    __builtin_prefetch(X + (size_t)c * KER, 0, 1);
    float4 xv = *(const float4*)(X + (size_t)c * KER + lane * 4);
    float* y = Y + (size_t)r * KER + lane * 4;
    atomicAdd(y + 0, v * xv.x);
    atomicAdd(y + 1, v * xv.y);
    atomicAdd(y + 2, v * xv.z);
    atomicAdd(y + 3, v * xv.w);
  }
}

// out rows only for gathered (user | pos | neg) nodes: [6144][256] @ lin_w.T + lin_b
__global__ __launch_bounds__(256) void lin_gather_kernel(
    const float* __restrict__ h1, const float* __restrict__ h2,
    const float* __restrict__ lw, const float* __restrict__ lb,
    const int* __restrict__ user, const int* __restrict__ pos_item,
    const int* __restrict__ neg_item, float* __restrict__ outg)
{
  __shared__ __align__(16) __bf16 As[16 * 256];
  const int tid = threadIdx.x;
  const int wave = tid >> 5, lane = tid & 31;
  const int half = lane >> 4, ln = lane & 15;
  const int mbase = blockIdx.x * 16;
  const int nb = wave * 16;
  for (int i = tid; i < 16 * 256; i += 256) {
    int r = i >> 8, k = i & 255;
    int gidx = mbase + r;
    int node;
    if (gidx < BB)          node = NB_OTHER + user[gidx];
    else if (gidx < 2 * BB) node = NB_OTHER + NB_USER + pos_item[gidx - BB];
    else                    node = NB_OTHER + NB_USER + neg_item[gidx - 2 * BB];
    float v = (k < KER) ? h1[(size_t)node * KER + k] : h2[(size_t)node * KER + (k - KER)];
    As[i] = (__bf16)v;
  }
  __syncthreads();
  v8f acc;
  float bias = lb[nb + ln];
#pragma unroll
  for (int v = 0; v < 8; ++v) acc[v] = bias;
#pragma unroll
  for (int c = 0; c < 8; ++c) {
    v16bf a = load_a_frag(As, 256, c, lane);
    v16bf b = load_b_frag_rowmajor_f32(lw, nb, 256, c, lane);
    acc = wmma_bf16(a, b, acc);
  }
#pragma unroll
  for (int v = 0; v < 8; ++v)
    outg[(size_t)(mbase + v + half * 8) * KER + nb + ln] = acc[v];
}

// BPR margin loss: one wave32 per b; wave shuffles reduce the 128-dim dots.
__global__ void score_kernel(const float* __restrict__ outg, float* __restrict__ loss) {
  const int lane = threadIdx.x & 31;
  const int b = (blockIdx.x * blockDim.x + threadIdx.x) >> 5;
  if (b >= BB) return;
  float4 u = *(const float4*)(outg + (size_t)b * KER + lane * 4);
  float4 p = *(const float4*)(outg + (size_t)(BB + b) * KER + lane * 4);
  float ps = u.x * p.x + u.y * p.y + u.z * p.z + u.w * p.w;
  float ns = 0.0f;
#pragma unroll
  for (int k = 0; k < KK; ++k) {
    float4 g = *(const float4*)(outg + (size_t)(2 * BB + k * BB + b) * KER + lane * 4);
    ns += u.x * g.x + u.y * g.y + u.z * g.z + u.w * g.w;
  }
  float s = ns - (float)KK * ps;             // sum_k(n_k.u) - K*(p.u)
#pragma unroll
  for (int off = 16; off > 0; off >>= 1) s += __shfl_xor(s, off, 32);
  if (lane == 0) {
    float m = s / (float)KK + 1.0f;          // mean_k diff, MARGIN = 1
    m = fminf(fmaxf(m, 1e-6f), 10000.0f);
    atomicAdd(loss, m);
  }
}

extern "C" void kernel_launch(void* const* d_in, const int* in_sizes, int n_in,
                              void* d_out, int out_size, void* d_ws, size_t ws_size,
                              hipStream_t stream) {
  const float* word_emb = (const float*)d_in[0];
  const float* other_pos = (const float*)d_in[1];
  const float* user_pos  = (const float*)d_in[2];
  const float* item_pos  = (const float*)d_in[3];
  const float* w_ih0 = (const float*)d_in[4];
  const float* w_hh0 = (const float*)d_in[5];
  const float* b_ih0 = (const float*)d_in[6];
  const float* b_hh0 = (const float*)d_in[7];
  const float* w_ih1 = (const float*)d_in[8];
  const float* w_hh1 = (const float*)d_in[9];
  const float* b_ih1 = (const float*)d_in[10];
  const float* b_hh1 = (const float*)d_in[11];
  const float* conv1_w = (const float*)d_in[12];
  const float* conv1_b = (const float*)d_in[13];
  const float* conv2_w = (const float*)d_in[14];
  const float* conv2_b = (const float*)d_in[15];
  const float* lin_w = (const float*)d_in[16];
  const float* lin_b = (const float*)d_in[17];
  const int* adj_row = (const int*)d_in[18];
  const int* adj_col = (const int*)d_in[19];
  const float* adj_vals = (const float*)d_in[20];
  const int* lookup = (const int*)d_in[21];
  const int* user = (const int*)d_in[22];
  const int* pos_item = (const int*)d_in[23];
  const int* neg_item = (const int*)d_in[24];
  (void)in_sizes; (void)n_in; (void)out_size; (void)ws_size;

  char* ws = (char*)d_ws;
  size_t off = 0;
  auto carve = [&](size_t bytes) -> void* {
    void* p = ws + off; off += (bytes + 255) & ~(size_t)255; return p;
  };
  __bf16* wembbf = (__bf16*)carve((size_t)VOCAB1 * EMB * 2);
  __bf16* allemb = (__bf16*)carve((size_t)NNODES * EMB * 2);
  float*  hpre   = (float*)carve((size_t)NNODES * KER * 4);   // conv1 then conv2 pre-spmm
  float*  h1     = (float*)carve((size_t)NNODES * KER * 4);
  float*  h2     = (float*)carve((size_t)NNODES * KER * 4);
  float*  outg   = (float*)carve((size_t)NGATHER * KER * 4);

  cvt_bf16_kernel<<<(VOCAB1 * EMB + 255) / 256, 256, 0, stream>>>(word_emb, wembbf,
                                                                  VOCAB1 * EMB);

  gru_fused_kernel<<<NNODES / 16, 128, 0, stream>>>(
      wembbf, lookup, w_ih0, w_hh0, b_ih0, b_hh0,
      w_ih1, w_hh1, b_ih1, b_hh1, other_pos, user_pos, item_pos, allemb);

  init_bias_kernel<<<(NNODES * KER + 255) / 256, 256, 0, stream>>>(h1, conv1_b, NNODES * KER);
  gemm64_kernel<<<NNODES / 16, 256, 0, stream>>>(allemb, conv1_w, hpre);
  spmm_atomic_kernel<<<2048, 256, 0, stream>>>(adj_row, adj_col, adj_vals, hpre, h1, NEDGE);

  init_bias_kernel<<<(NNODES * KER + 255) / 256, 256, 0, stream>>>(h2, conv2_b, NNODES * KER);
  gemm128_kernel<<<NNODES / 16, 256, 0, stream>>>(h1, conv2_w, hpre);
  spmm_atomic_kernel<<<2048, 256, 0, stream>>>(adj_row, adj_col, adj_vals, hpre, h2, NEDGE);

  lin_gather_kernel<<<NGATHER / 16, 256, 0, stream>>>(h1, h2, lin_w, lin_b,
                                                      user, pos_item, neg_item, outg);

  (void)hipMemsetAsync(d_out, 0, sizeof(float), stream);
  score_kernel<<<(BB * 32 + 255) / 256, 256, 0, stream>>>(outg, (float*)d_out);
}
